// MLAttention_936302871130
// MI455X (gfx1250) — matrix operand order repaired and checked by reading, not compiled, not run on hardware
//
#include <hip/hip_runtime.h>
#include <hip/hip_bf16.h>
#include <cstdint>

#define NUM_HEADS  16
#define HEAD_DIM   128
#define HIDDEN_DIM 2048
#define CKV_DIM    512
#define CQ_DIM     1024
#define ROPE_D     64
#define BATCH      2
#define SEQ        2048
#define TOKENS     (BATCH * SEQ)   // 4096

typedef __attribute__((ext_vector_type(16))) __bf16 v16bf;
typedef __attribute__((ext_vector_type(8)))  __bf16 v8bf;
typedef __attribute__((ext_vector_type(8)))  float  v8f;
typedef __attribute__((ext_vector_type(4)))  int    v4i;

// ---- CDNA5 async copy to LDS (ASYNCcnt path), guarded so compile stays green ----
#if defined(__has_builtin)
#if __has_builtin(__builtin_amdgcn_global_load_async_to_lds_b128) && \
    __has_builtin(__builtin_amdgcn_s_wait_asynccnt)
#define USE_ASYNC_LDS 1
#endif
#endif
#ifndef USE_ASYNC_LDS
#define USE_ASYNC_LDS 0
#endif

#if USE_ASYNC_LDS
// builtin signature (from compiler diagnostic): (v4i addrspace(1)*, v4i addrspace(3)*, imm offset, imm cpol)
#define ASYNC_CP16B(gsrc, ldst)                                                   \
  __builtin_amdgcn_global_load_async_to_lds_b128(                                 \
      (__attribute__((address_space(1))) v4i*)(gsrc),                             \
      (__attribute__((address_space(3))) v4i*)(ldst), 0, 0)
#endif

__device__ __forceinline__ unsigned short f32_to_bf16(float f) {
  unsigned int u = __float_as_uint(f);
  u += 0x7fffu + ((u >> 16) & 1u);           // round-to-nearest-even
  return (unsigned short)(u >> 16);
}
__device__ __forceinline__ float bf16_to_f32(unsigned short h) {
  return __uint_as_float(((unsigned int)h) << 16);
}
__device__ __forceinline__ v8f zero8() {
  v8f z;
#pragma unroll
  for (int i = 0; i < 8; ++i) z[i] = 0.0f;
  return z;
}

// A-fragment (16x32 bf16): lane(m = lane&15, half = lane>>4),
// elem e = A[m][k0 + half*16 + e] (contiguous order, matches load_b16)
__device__ __forceinline__ v16bf load_a_frag(const unsigned short* base, int stride,
                                             int m, int k0, int half) {
  v16bf a;
  const unsigned short* p = base + (size_t)m * stride + k0 + half * 16;
  *(v8bf*)&a       = *(const v8bf*)p;
  *((v8bf*)&a + 1) = *(const v8bf*)(p + 8);
  return a;
}
// B-fragment (32x16 bf16): lane(n = lane&15, half), elem e = B[half*16+e][n].
// p points at 16 contiguous contract-dim elements of column n:
__device__ __forceinline__ v16bf load_b16(const unsigned short* p) {
  v16bf b;
  *(v8bf*)&b       = *(const v8bf*)p;
  *((v8bf*)&b + 1) = *(const v8bf*)(p + 8);
  return b;
}
__device__ __forceinline__ v8f wmma_bf16(v16bf a, v16bf b, v8f c) {
  return __builtin_amdgcn_wmma_f32_16x16x32_bf16(false, a, false, b, (short)0, c, false, false);
}

// ---------------- elementwise kernels ----------------
__global__ void convert_f32_bf16(const float* __restrict__ x,
                                 unsigned short* __restrict__ y, int n) {
  int i = blockIdx.x * blockDim.x + threadIdx.x;
  if (i < n) y[i] = f32_to_bf16(x[i]);
}

// fused convert + transpose: W (K x N) fp32  ->  Wt (N x K) bf16
__global__ void convert_transpose_bf16(const float* __restrict__ W,
                                       unsigned short* __restrict__ Wt,
                                       int K, int N) {
  int idx = blockIdx.x * blockDim.x + threadIdx.x;
  if (idx >= K * N) return;
  int n = idx % N, k = idx / N;          // n fast -> coalesced reads
  Wt[(size_t)n * K + k] = f32_to_bf16(W[(size_t)k * N + n]);
}

__global__ void rope_q_kernel(unsigned short* __restrict__ qr) {
  // qr: (TOKENS, NUM_HEADS, ROPE_D); one thread per rotation pair
  int idx = blockIdx.x * blockDim.x + threadIdx.x;
  if (idx >= TOKENS * NUM_HEADS * (ROPE_D / 2)) return;
  int i = idx & 31; int rest = idx >> 5;
  int h = rest & (NUM_HEADS - 1); int tok = rest >> 4;
  int s = tok & (SEQ - 1);
  float inv = __powf(10000.0f, -(float)(2 * i) / (float)ROPE_D);
  float sn, cs; __sincosf((float)s * inv, &sn, &cs);
  unsigned short* p = qr + (size_t)tok * (NUM_HEADS * ROPE_D) + h * ROPE_D;
  float x1 = bf16_to_f32(p[i]), x2 = bf16_to_f32(p[i + 32]);
  p[i]      = f32_to_bf16(x1 * cs - x2 * sn);
  p[i + 32] = f32_to_bf16(x2 * cs + x1 * sn);
}

__global__ void rope_k_kernel(unsigned short* __restrict__ kr) {
  // kr: (TOKENS, ROPE_D)
  int idx = blockIdx.x * blockDim.x + threadIdx.x;
  if (idx >= TOKENS * (ROPE_D / 2)) return;
  int i = idx & 31; int tok = idx >> 5;
  int s = tok & (SEQ - 1);
  float inv = __powf(10000.0f, -(float)(2 * i) / (float)ROPE_D);
  float sn, cs; __sincosf((float)s * inv, &sn, &cs);
  unsigned short* p = kr + (size_t)tok * ROPE_D;
  float x1 = bf16_to_f32(p[i]), x2 = bf16_to_f32(p[i + 32]);
  p[i]      = f32_to_bf16(x1 * cs - x2 * sn);
  p[i + 32] = f32_to_bf16(x2 * cs + x1 * sn);
}

__global__ void transpose_v_kernel(const unsigned short* __restrict__ v,
                                   unsigned short* __restrict__ vt) {
  // v: (B*S, NH*HD) -> vt: (B, NH, HD, S)
  int idx = blockIdx.x * blockDim.x + threadIdx.x;
  if (idx >= BATCH * SEQ * NUM_HEADS * HEAD_DIM) return;
  int d = idx & (HEAD_DIM - 1); int rest = idx >> 7;
  int h = rest & (NUM_HEADS - 1); rest >>= 4;
  int s = rest & (SEQ - 1); int b = rest >> 11;
  vt[(((size_t)b * NUM_HEADS + h) * HEAD_DIM + d) * SEQ + s] =
      v[((size_t)b * SEQ + s) * (NUM_HEADS * HEAD_DIM) + h * HEAD_DIM + d];
}

// ---------------- tiled bf16 GEMM: C(MxN) = A(MxK) * Bt(NxK)^T ----------------
// Block tile 128x128, BK=32, 8 waves in 4x2; each wave owns a 32x64 sub-tile
// (2 A-frags x 4 B-frags -> 8 WMMAs per k-step). Double-buffered LDS with
// async copies overlapping WMMA: wait asynccnt<=4 keeps the in-flight tile moving.
template <bool OUT_F32>
__global__ __launch_bounds__(256) void gemm_bf16(const unsigned short* __restrict__ A,
                                                 const unsigned short* __restrict__ Bt,
                                                 void* __restrict__ C,
                                                 int M, int N, int K) {
  __shared__ unsigned short As[2][128 * 32];   // [buf][row][k]
  __shared__ unsigned short Bs[2][128 * 32];   // [buf][n][k]
  const int tid  = threadIdx.x;
  const int lane = tid & 31;
  const int wave = tid >> 5;
  const int wm   = wave >> 1;               // 0..3 -> rows wm*32..+31
  const int wn   = wave & 1;                // 0..1 -> cols wn*64..+63
  const int m16  = lane & 15, half = lane >> 4;
  const int mBase = blockIdx.y * 128;
  const int nBase = blockIdx.x * 128;

  const int srow = tid >> 1;                // 0..127 staging row
  const int scol = (tid & 1) * 16;          // 0 or 16
  // OOB column guard by source-row clamp: columns >= N are computed on junk
  // and discarded at the epilogue (each output column uses only its own B row).
  const int bRow = (nBase + srow < N) ? (nBase + srow) : (N - 1);
  const unsigned short* gAr = A  + (size_t)(mBase + srow) * K + scol;
  const unsigned short* gBr = Bt + (size_t)bRow * K + scol;

  auto stage = [&](int buf, int k0) {
#if USE_ASYNC_LDS
    ASYNC_CP16B(gAr + k0,     &As[buf][srow * 32 + scol]);
    ASYNC_CP16B(gAr + k0 + 8, &As[buf][srow * 32 + scol + 8]);
    ASYNC_CP16B(gBr + k0,     &Bs[buf][srow * 32 + scol]);
    ASYNC_CP16B(gBr + k0 + 8, &Bs[buf][srow * 32 + scol + 8]);
#else
    *(v8bf*)&As[buf][srow * 32 + scol]     = *(const v8bf*)(gAr + k0);
    *(v8bf*)&As[buf][srow * 32 + scol + 8] = *(const v8bf*)(gAr + k0 + 8);
    *(v8bf*)&Bs[buf][srow * 32 + scol]     = *(const v8bf*)(gBr + k0);
    *(v8bf*)&Bs[buf][srow * 32 + scol + 8] = *(const v8bf*)(gBr + k0 + 8);
#endif
  };

  v8f acc[2][4];
#pragma unroll
  for (int mt = 0; mt < 2; ++mt)
#pragma unroll
    for (int t = 0; t < 4; ++t) acc[mt][t] = zero8();

  stage(0, 0);                               // prologue: tile 0 -> buffer 0
  int buf = 0;
  for (int k0 = 0; k0 < K; k0 += 32, buf ^= 1) {
    const bool more = (k0 + 32 < K);
    if (more) {                              // prefetch tile k+2 into cache
      __builtin_prefetch(gAr + k0 + 64, 0, 1);
      __builtin_prefetch(gBr + k0 + 64, 0, 1);
    }
    // stage next tile into the other buffer (wrap to dead copies on last iter
    // so the per-wave ASYNCcnt stays uniform for the wait threshold below)
    stage(buf ^ 1, more ? k0 + 32 : 0);
#if USE_ASYNC_LDS
    __builtin_amdgcn_s_wait_asynccnt(4);     // in-order: current tile has landed
#endif
    __syncthreads();
    v16bf a0 = load_a_frag(&As[buf][0], 32, wm * 32 + m16,      0, half);
    v16bf a1 = load_a_frag(&As[buf][0], 32, wm * 32 + 16 + m16, 0, half);
#pragma unroll
    for (int t = 0; t < 4; ++t) {
      v16bf b = load_b16(&Bs[buf][(wn * 64 + t * 16 + m16) * 32 + half * 16]);
      acc[0][t] = wmma_bf16(a0, b, acc[0][t]);
      acc[1][t] = wmma_bf16(a1, b, acc[1][t]);
    }
    __syncthreads();                         // readers done before buf is re-staged
  }
  // C/D layout: lane(n = lane&15, half), VGPR r -> row half*8+r
#pragma unroll
  for (int mt = 0; mt < 2; ++mt)
#pragma unroll
    for (int t = 0; t < 4; ++t) {
      size_t col = (size_t)(nBase + wn * 64 + t * 16 + m16);
      if (col >= (size_t)N) continue;
#pragma unroll
      for (int r = 0; r < 8; ++r) {
        size_t row = (size_t)(mBase + wm * 32 + mt * 16 + half * 8 + r);
        if (OUT_F32) ((float*)C)[row * N + col] = acc[mt][t][r];
        else ((unsigned short*)C)[row * N + col] = f32_to_bf16(acc[mt][t][r]);
      }
    }
}

// ---------------- flash attention (1 wave per (b, h, 16-row q tile)) ----------------
__global__ __launch_bounds__(32) void mla_attention(
    const unsigned short* __restrict__ qc,  // (B*S, NH*HD)
    const unsigned short* __restrict__ qr,  // (B*S, NH*ROPE_D), rope applied
    const unsigned short* __restrict__ kc,  // (B*S, NH*HD)
    const unsigned short* __restrict__ kr,  // (B*S, ROPE_D), rope applied
    const unsigned short* __restrict__ vt,  // (B, NH, HD, S)
    unsigned short* __restrict__ attn) {    // (B*S, NH*HD)
  __shared__ unsigned short Plds[16 * 32];
  const int lane = threadIdx.x;
  const int m16 = lane & 15, half = lane >> 4;
  int bid = blockIdx.x;
  const int qt = bid & 127; bid >>= 7;
  const int h  = bid & (NUM_HEADS - 1); bid >>= 4;
  const int b  = bid;
  const int qbase = qt * 16;
  const size_t tokQ = (size_t)b * SEQ + qbase;
  const float scale = 0.07216878364870323f;   // 1/sqrt(192)

  // preload q fragments: 4 chunks of q_C (128 dims) + 2 chunks of q_R (64 dims)
  v16bf qf[6];
#pragma unroll
  for (int c = 0; c < 4; ++c)
    qf[c] = load_a_frag(qc + tokQ * (NUM_HEADS * HEAD_DIM) + h * HEAD_DIM,
                        NUM_HEADS * HEAD_DIM, m16, c * 32, half);
#pragma unroll
  for (int c = 0; c < 2; ++c)
    qf[4 + c] = load_a_frag(qr + tokQ * (NUM_HEADS * ROPE_D) + h * ROPE_D,
                            NUM_HEADS * ROPE_D, m16, c * 32, half);

  v8f acc[8];
#pragma unroll
  for (int t = 0; t < 8; ++t) acc[t] = zero8();
  float mrow[8], lrow[8];
#pragma unroll
  for (int r = 0; r < 8; ++r) { mrow[r] = -1e30f; lrow[r] = 0.0f; }

  const unsigned short* vbase = vt + (((size_t)b * NUM_HEADS + h) * HEAD_DIM) * SEQ;

  for (int kb = 0; kb <= qbase; kb += 32) {   // 32-key chunks, causal bound
    if (kb + 32 <= qbase) {                   // prefetch next chunk's K/V stream
      __builtin_prefetch(kc + ((size_t)b * SEQ + kb + 32 + m16 * 2) * (NUM_HEADS * HEAD_DIM)
                             + h * HEAD_DIM, 0, 1);
      __builtin_prefetch(vbase + (size_t)(lane * 4) * SEQ + kb + 32, 0, 1);
    }
    v8f s[2]; s[0] = zero8(); s[1] = zero8();
#pragma unroll
    for (int j = 0; j < 2; ++j) {
      const size_t tokK = (size_t)b * SEQ + kb + j * 16 + m16;
#pragma unroll
      for (int c = 0; c < 4; ++c) {           // content part: B-frag = rows of k_C
        v16bf kf = load_b16(kc + tokK * (NUM_HEADS * HEAD_DIM) + h * HEAD_DIM + c * 32 + half * 16);
        s[j] = wmma_bf16(qf[c], kf, s[j]);
      }
#pragma unroll
      for (int c = 0; c < 2; ++c) {           // rope part (shared across heads)
        v16bf kf = load_b16(kr + tokK * ROPE_D + c * 32 + half * 16);
        s[j] = wmma_bf16(qf[4 + c], kf, s[j]);
      }
    }
    // scale + causal mask (reference adds -1e9 on masked)
#pragma unroll
    for (int j = 0; j < 2; ++j) {
      const int key = kb + j * 16 + m16;
#pragma unroll
      for (int r = 0; r < 8; ++r) {
        int q = qbase + half * 8 + r;
        float v = s[j][r] * scale;
        s[j][r] = (key > q) ? (v - 1e9f) : v;
      }
    }
    // online softmax; row reductions stay within each 16-lane half (wave32)
    float alpha8[8];
#pragma unroll
    for (int r = 0; r < 8; ++r) {
      float mx = fmaxf(s[0][r], s[1][r]);
#pragma unroll
      for (int o = 8; o >= 1; o >>= 1) mx = fmaxf(mx, __shfl_xor(mx, o, 32));
      float mn = fmaxf(mrow[r], mx);
      float al = __expf(mrow[r] - mn);
      mrow[r] = mn;
      float p0 = __expf(s[0][r] - mn);
      float p1 = __expf(s[1][r] - mn);
      s[0][r] = p0; s[1][r] = p1;
      float rs = p0 + p1;
#pragma unroll
      for (int o = 8; o >= 1; o >>= 1) rs += __shfl_xor(rs, o, 32);
      lrow[r] = lrow[r] * al + rs;
      alpha8[r] = al;
    }
#pragma unroll
    for (int t = 0; t < 8; ++t)
#pragma unroll
      for (int r = 0; r < 8; ++r) acc[t][r] *= alpha8[r];

    // probs: C-layout -> LDS [m][k] (same-wave LDS ops are in order)
#pragma unroll
    for (int j = 0; j < 2; ++j)
#pragma unroll
      for (int r = 0; r < 8; ++r)
        Plds[(half * 8 + r) * 32 + j * 16 + m16] = f32_to_bf16(s[j][r]);
    v16bf pf = load_a_frag(Plds, 32, m16, 0, half);

    // PV: B-frag = columns of V = contiguous rows of Vt
#pragma unroll
    for (int t = 0; t < 8; ++t) {
      v16bf vf = load_b16(vbase + (size_t)(t * 16 + m16) * SEQ + kb + half * 16);
      acc[t] = wmma_bf16(pf, vf, acc[t]);
    }
  }
  // normalize and store
#pragma unroll
  for (int t = 0; t < 8; ++t)
#pragma unroll
    for (int r = 0; r < 8; ++r) {
      size_t row = tokQ + half * 8 + r;
      attn[row * (NUM_HEADS * HEAD_DIM) + h * HEAD_DIM + t * 16 + m16] =
          f32_to_bf16(acc[t][r] / lrow[r]);
    }
}

// ---------------- launcher ----------------
extern "C" void kernel_launch(void* const* d_in, const int* in_sizes, int n_in,
                              void* d_out, int out_size, void* d_ws, size_t ws_size,
                              hipStream_t stream) {
  (void)in_sizes; (void)n_in; (void)out_size; (void)ws_size;
  const float* hidden = (const float*)d_in[0];
  // d_in[1] = mask (causal, computed analytically in-kernel)
  const float* W_DKV = (const float*)d_in[2];
  const float* W_UK  = (const float*)d_in[3];
  const float* W_UV  = (const float*)d_in[4];
  const float* W_DQ  = (const float*)d_in[5];
  const float* W_UQ  = (const float*)d_in[6];
  const float* W_QR  = (const float*)d_in[7];
  const float* W_KR  = (const float*)d_in[8];
  const float* W_O   = (const float*)d_in[9];
  float* out = (float*)d_out;

  char* ws = (char*)d_ws;
  size_t off = 0;
  auto alloc_bf = [&](size_t elems) -> unsigned short* {
    unsigned short* p = (unsigned short*)(ws + off);
    off = (off + elems * 2 + 255) & ~(size_t)255;
    return p;
  };
  unsigned short* hid_bf  = alloc_bf((size_t)TOKENS * HIDDEN_DIM);
  // pre-transposed bf16 weights: Wt (N x K)
  unsigned short* wdkv_t = alloc_bf((size_t)HIDDEN_DIM * CKV_DIM);
  unsigned short* wuk_t  = alloc_bf((size_t)CKV_DIM * NUM_HEADS * HEAD_DIM);
  unsigned short* wuv_t  = alloc_bf((size_t)CKV_DIM * NUM_HEADS * HEAD_DIM);
  unsigned short* wdq_t  = alloc_bf((size_t)HIDDEN_DIM * CQ_DIM);
  unsigned short* wuq_t  = alloc_bf((size_t)CQ_DIM * NUM_HEADS * HEAD_DIM);
  unsigned short* wqr_t  = alloc_bf((size_t)CQ_DIM * NUM_HEADS * ROPE_D);
  unsigned short* wkr_t  = alloc_bf((size_t)HIDDEN_DIM * ROPE_D);
  unsigned short* wo_t   = alloc_bf((size_t)NUM_HEADS * HEAD_DIM * HIDDEN_DIM);
  unsigned short* ckv_bf  = alloc_bf((size_t)TOKENS * CKV_DIM);
  unsigned short* cq_bf   = alloc_bf((size_t)TOKENS * CQ_DIM);
  unsigned short* kc_bf   = alloc_bf((size_t)TOKENS * NUM_HEADS * HEAD_DIM);
  unsigned short* v_bf    = alloc_bf((size_t)TOKENS * NUM_HEADS * HEAD_DIM);
  unsigned short* qc_bf   = alloc_bf((size_t)TOKENS * NUM_HEADS * HEAD_DIM);
  unsigned short* qr_bf   = alloc_bf((size_t)TOKENS * NUM_HEADS * ROPE_D);
  unsigned short* kr_bf   = alloc_bf((size_t)TOKENS * ROPE_D);
  unsigned short* vt_bf   = alloc_bf((size_t)TOKENS * NUM_HEADS * HEAD_DIM);
  unsigned short* attn_bf = alloc_bf((size_t)TOKENS * NUM_HEADS * HEAD_DIM);

  convert_f32_bf16<<<(TOKENS * HIDDEN_DIM + 255) / 256, 256, 0, stream>>>(
      hidden, hid_bf, TOKENS * HIDDEN_DIM);
  auto convT = [&](const float* src, unsigned short* dst, int K, int N) {
    convert_transpose_bf16<<<(K * N + 255) / 256, 256, 0, stream>>>(src, dst, K, N);
  };
  convT(W_DKV, wdkv_t, HIDDEN_DIM, CKV_DIM);
  convT(W_UK,  wuk_t,  CKV_DIM, NUM_HEADS * HEAD_DIM);
  convT(W_UV,  wuv_t,  CKV_DIM, NUM_HEADS * HEAD_DIM);
  convT(W_DQ,  wdq_t,  HIDDEN_DIM, CQ_DIM);
  convT(W_UQ,  wuq_t,  CQ_DIM, NUM_HEADS * HEAD_DIM);
  convT(W_QR,  wqr_t,  CQ_DIM, NUM_HEADS * ROPE_D);
  convT(W_KR,  wkr_t,  HIDDEN_DIM, ROPE_D);
  convT(W_O,   wo_t,   NUM_HEADS * HEAD_DIM, HIDDEN_DIM);

  dim3 blk(256);
  auto grid = [](int M, int N) { return dim3((N + 127) / 128, M / 128); };
  // projections
  gemm_bf16<false><<<grid(TOKENS, CKV_DIM), blk, 0, stream>>>(hid_bf, wdkv_t, ckv_bf, TOKENS, CKV_DIM, HIDDEN_DIM);
  gemm_bf16<false><<<grid(TOKENS, CQ_DIM),  blk, 0, stream>>>(hid_bf, wdq_t,  cq_bf,  TOKENS, CQ_DIM,  HIDDEN_DIM);
  gemm_bf16<false><<<grid(TOKENS, NUM_HEADS*HEAD_DIM), blk, 0, stream>>>(ckv_bf, wuk_t, kc_bf, TOKENS, NUM_HEADS*HEAD_DIM, CKV_DIM);
  gemm_bf16<false><<<grid(TOKENS, NUM_HEADS*HEAD_DIM), blk, 0, stream>>>(ckv_bf, wuv_t, v_bf,  TOKENS, NUM_HEADS*HEAD_DIM, CKV_DIM);
  gemm_bf16<false><<<grid(TOKENS, NUM_HEADS*HEAD_DIM), blk, 0, stream>>>(cq_bf,  wuq_t, qc_bf, TOKENS, NUM_HEADS*HEAD_DIM, CQ_DIM);
  gemm_bf16<false><<<grid(TOKENS, NUM_HEADS*ROPE_D),   blk, 0, stream>>>(cq_bf,  wqr_t, qr_bf, TOKENS, NUM_HEADS*ROPE_D,   CQ_DIM);
  gemm_bf16<false><<<grid(TOKENS, ROPE_D),             blk, 0, stream>>>(hid_bf, wkr_t, kr_bf, TOKENS, ROPE_D,             HIDDEN_DIM);
  // rope
  rope_q_kernel<<<(TOKENS * NUM_HEADS * 32 + 255) / 256, 256, 0, stream>>>(qr_bf);
  rope_k_kernel<<<(TOKENS * 32 + 255) / 256, 256, 0, stream>>>(kr_bf);
  // V^T for PV fragment loads
  transpose_v_kernel<<<(TOKENS * NUM_HEADS * HEAD_DIM + 255) / 256, 256, 0, stream>>>(v_bf, vt_bf);
  // attention: one wave per (b, h, 16-row q tile)
  mla_attention<<<BATCH * NUM_HEADS * (SEQ / 16), 32, 0, stream>>>(qc_bf, qr_bf, kc_bf, kr_bf, vt_bf, attn_bf);
  // output projection -> fp32
  gemm_bf16<true><<<grid(TOKENS, HIDDEN_DIM), blk, 0, stream>>>(attn_bf, wo_t, out, TOKENS, HIDDEN_DIM, NUM_HEADS * HEAD_DIM);
}
